// SASRecModel_78013785964782
// MI455X (gfx1250) — compile-verified, matrix operand order checked by reading, not run on hardware
//
#include <hip/hip_runtime.h>

// ---------------------------------------------------------------------------
// Shapes (fixed by the reference)
// ---------------------------------------------------------------------------
#define HID 128
#define DTX 768
#define DIM 512
#define NEXP 8
#define LSEQ 200
#define NTOK (128 * 200)   // B*L = 25600
#define TILE 16            // tokens per workgroup

typedef __attribute__((ext_vector_type(16))) __bf16 v16bf;
typedef __attribute__((ext_vector_type(8)))  __bf16 v8bf;
typedef __attribute__((ext_vector_type(4)))  __bf16 v4bf;
typedef __attribute__((ext_vector_type(8)))  float  v8f;
typedef __attribute__((ext_vector_type(4)))  float  v4f;

union VF { v8f v; float f[8]; };

__device__ __forceinline__ v4bf cvt4(v4f d) {
    v4bf b = { (__bf16)d.x, (__bf16)d.y, (__bf16)d.z, (__bf16)d.w };
    return b;
}

// Load 16 contiguous bf16 (one WMMA fragment slice) as two b128 loads.
__device__ __forceinline__ v16bf ld_frag(const __bf16* p) {
    union { v16bf v; v8bf h[2]; } u;
    u.h[0] = *(const v8bf*)(p);
    u.h[1] = *(const v8bf*)(p + 8);
    return u.v;
}

// C[16x16] += A[16xKD] * B[KD x 16]
// A: LDS, row-major [16][lda]; B: global bf16 weights, "row n = output col n",
// row-major [*][ldb] with the 16-col base already applied by the caller.
template <int KD>
__device__ __forceinline__ v8f bf16_gemm(const __bf16* __restrict__ As, int lda,
                                         const __bf16* __restrict__ Bw, int ldb,
                                         int lane, v8f acc) {
    const int r  = lane & 15;          // A-row / B-col inside the tile
    const int ko = (lane >> 4) * 16;   // K half selected by lane group
    const __bf16* ap = As + r * lda + ko;
    const __bf16* bp = Bw + r * ldb + ko;
#pragma unroll
    for (int k = 0; k < KD; k += 32) {
        v16bf a = ld_frag(ap + k);
        v16bf b = ld_frag(bp + k);
        acc = __builtin_amdgcn_wmma_f32_16x16x32_bf16(
            /*neg_a=*/false, a, /*neg_b=*/false, b,
            /*c_mod=*/(short)0, acc, /*reuse_a=*/false, /*reuse_b=*/false);
    }
    return acc;
}

// Scatter a C tile (+ per-column bias) into an LDS f32 [16][128] buffer.
__device__ __forceinline__ void store_tile_bias(float* dst, v8f c, float bias,
                                                int lane, int nb) {
    VF u; u.v = c;
    const int n  = nb + (lane & 15);
    const int mb = (lane >> 4) * 8;
#pragma unroll
    for (int i = 0; i < 8; i++) dst[(mb + i) * HID + n] = u.f[i] + bias;
}

// softmax gate -> top-2 -> renormalize (one row, scalar FP32)
__device__ __forceinline__ void route_row(const float* zrow, const float* gw,
                                          const float* gb, float* grow) {
    float lg[NEXP];
#pragma unroll
    for (int e = 0; e < NEXP; e++) {
        float d = gb[e];
        for (int h = 0; h < HID; h++) d += zrow[h] * gw[e * HID + h];
        lg[e] = d;
    }
    float mx = lg[0];
#pragma unroll
    for (int e = 1; e < NEXP; e++) mx = fmaxf(mx, lg[e]);
    float sum = 0.f;
#pragma unroll
    for (int e = 0; e < NEXP; e++) { lg[e] = __expf(lg[e] - mx); sum += lg[e]; }
    const float inv = 1.f / sum;
#pragma unroll
    for (int e = 0; e < NEXP; e++) lg[e] *= inv;
    int i1 = 0;
#pragma unroll
    for (int e = 1; e < NEXP; e++) if (lg[e] > lg[i1]) i1 = e;
    int i2 = (i1 == 0) ? 1 : 0;
#pragma unroll
    for (int e = 0; e < NEXP; e++) if (e != i1 && lg[e] > lg[i2]) i2 = e;
    const float s2 = 1.f / (lg[i1] + lg[i2] + 1e-8f);
#pragma unroll
    for (int e = 0; e < NEXP; e++)
        grow[e] = (e == i1 || e == i2) ? lg[e] * s2 : 0.f;
}

// ---------------------------------------------------------------------------
// FP32 -> BF16 weight repack (vectorized: all weight sizes are /4)
// ---------------------------------------------------------------------------
__global__ void cvt_bf16_kernel(const float* __restrict__ src,
                                __bf16* __restrict__ dst, int n4) {
    int i = blockIdx.x * 256 + threadIdx.x;
    if (i < n4) ((v4bf*)dst)[i] = cvt4(((const v4f*)src)[i]);
}

// ---------------------------------------------------------------------------
// Fused SASRec multimodal-MoE kernel
// ---------------------------------------------------------------------------
struct Params {
    const int*   ids;
    const float* t_noise; const float* i_noise;
    const float* item_table; const float* pos_table;
    const float* text_table; const float* img_table;
    const float* fc_text_b; const float* fc_img_b;
    const float* ln_w; const float* ln_b;
    const float* mu_t_b; const float* sg_t_b;
    const float* mu_i_b; const float* sg_i_b;
    const float* gate_w; const float* gate_b;
    const float* texp_b; const float* iexp_b;
    const float* fus_b; const float* fus_ln_w; const float* fus_ln_b;
    const __bf16 *w_text, *w_img, *w_mu_t, *w_sg_t, *w_mu_i, *w_sg_i;
    const __bf16 *w_texp, *w_iexp, *w_fus;
    float* out;
};

__global__ __launch_bounds__(256)
void fused_mm_moe_kernel(Params p) {
    __shared__ struct {
        int   ids[TILE];
        float inv[TILE];
        float gate[TILE][NEXP];
        alignas(16) __bf16 bufA[TILE * DTX];   // text rows / img rows / cat[16][256]
        alignas(16) float  seq[TILE][HID];
        alignas(16) float  x[TILE][HID];       // proj / fusion f32 scratch
        alignas(16) __bf16 xbf[TILE][HID];
        alignas(16) float  z[TILE][HID];
        alignas(16) __bf16 zbf[TILE][HID];
    } s;

    const int tid  = threadIdx.x;
    const int lane = tid & 31;
    const int nb   = (tid >> 5) * 16;     // this wave's 16-col output base
    const int t0   = blockIdx.x * TILE;

    // ---- phase 1: token ids, raw seq emb, text rows (bf16 in LDS) ----------
    if (tid < TILE) s.ids[tid] = p.ids[t0 + tid];
    __syncthreads();
    // seq = item_table[id] + pos_table[l]   (float4 loads, 512 vec4/block)
    for (int i = tid; i < TILE * HID / 4; i += 256) {
        const int m = i >> 5, h4 = i & 31;
        const int tok = t0 + m;
        const v4f a = ((const v4f*)(p.item_table + s.ids[m] * HID))[h4];
        const v4f b = ((const v4f*)(p.pos_table + (tok % LSEQ) * HID))[h4];
        ((v4f*)&s.seq[0][0])[i] = a + b;
    }
    // text rows -> bf16 in LDS (float4 loads, packed b64 LDS stores)
    for (int i = tid; i < TILE * DTX / 4; i += 256) {
        const int m = i / (DTX / 4), c4 = i - m * (DTX / 4);
        const v4f d = ((const v4f*)(p.text_table + (long)s.ids[m] * DTX))[c4];
        ((v4bf*)s.bufA)[i] = cvt4(d);
    }
    __syncthreads();

    // seq LayerNorm (eps 1e-12)
    if (tid < TILE) {
        float mean = 0.f;
        for (int h = 0; h < HID; h++) mean += s.seq[tid][h];
        mean *= (1.f / HID);
        float var = 0.f;
        for (int h = 0; h < HID; h++) { float d = s.seq[tid][h] - mean; var += d * d; }
        var *= (1.f / HID);
        const float rs = rsqrtf(var + 1e-12f);
        for (int h = 0; h < HID; h++)
            s.seq[tid][h] = (s.seq[tid][h] - mean) * rs * p.ln_w[h] + p.ln_b[h];
    }
    __syncthreads();

    VF acc_t, acc_i;

    // ======================= TEXT then IMAGE modality =======================
#pragma unroll 1
    for (int mod = 0; mod < 2; mod++) {
        const __bf16* w_pj  = mod ? p.w_img : p.w_text;
        const float*  b_pj  = mod ? p.fc_img_b : p.fc_text_b;
        const __bf16* w_mu  = mod ? p.w_mu_i : p.w_mu_t;
        const __bf16* w_sg  = mod ? p.w_sg_i : p.w_sg_t;
        const float*  b_mu  = mod ? p.mu_i_b : p.mu_t_b;
        const float*  b_sg  = mod ? p.sg_i_b : p.sg_t_b;
        const float*  noise = mod ? p.i_noise : p.t_noise;
        const __bf16* w_ex  = mod ? p.w_iexp : p.w_texp;
        const float*  b_ex  = mod ? p.iexp_b : p.texp_b;

        if (mod == 1) {
            // reload staging buffer with image rows (float4 / packed b64)
            for (int i = tid; i < TILE * DIM / 4; i += 256) {
                const int m = i >> 7, c4 = i & 127;
                const v4f d = ((const v4f*)(p.img_table + (long)s.ids[m] * DIM))[c4];
                ((v4bf*)s.bufA)[i] = cvt4(d);
            }
            __syncthreads();
        }

        // modality projection: [16 x kproj] @ [kproj x 128]
        {
            v8f c = {};
            if (mod == 0) c = bf16_gemm<DTX>(s.bufA, DTX, w_pj + nb * DTX, DTX, lane, c);
            else          c = bf16_gemm<DIM>(s.bufA, DIM, w_pj + nb * DIM, DIM, lane, c);
            store_tile_bias(&s.x[0][0], c, b_pj[nb + (lane & 15)], lane, nb);
        }
        __syncthreads();

        // l2norm scale per row
        if (tid < TILE) {
            float ss = 0.f;
            for (int h = 0; h < HID; h++) ss += s.x[tid][h] * s.x[tid][h];
            s.inv[tid] = 1.f / fmaxf(sqrtf(ss), 1e-12f);
        }
        __syncthreads();
        for (int i = tid; i < TILE * HID / 4; i += 256) {
            const int m = i >> 5;
            v4f d = ((const v4f*)&s.x[0][0])[i] * s.inv[m];
            ((v4bf*)&s.xbf[0][0])[i] = cvt4(d);
        }
        __syncthreads();

        // mu / sigma heads (both tiles in registers) -> VAE reparam z
        {
            v8f cm = {}, cs = {};
            cm = bf16_gemm<HID>(&s.xbf[0][0], HID, w_mu + nb * HID, HID, lane, cm);
            cs = bf16_gemm<HID>(&s.xbf[0][0], HID, w_sg + nb * HID, HID, lane, cs);
            const int n  = nb + (lane & 15);
            const int mb = (lane >> 4) * 8;
            const float bm = b_mu[n], bs = b_sg[n];
            VF um, us; um.v = cm; us.v = cs;
#pragma unroll
            for (int i = 0; i < 8; i++) {
                const int m = mb + i;
                const float zv = (um.f[i] + bm) +
                                 __expf(us.f[i] + bs) *
                                 noise[(long)(t0 + m) * HID + n];
                s.z[m][n]   = zv;
                s.zbf[m][n] = (__bf16)zv;
            }
        }
        __syncthreads();

        // top-2 softmax routing
        if (tid < TILE) route_row(s.z[tid], p.gate_w, p.gate_b, s.gate[tid]);
        __syncthreads();

        // dense experts, gate-weighted accumulation in registers
        VF acc;
#pragma unroll
        for (int i = 0; i < 8; i++) acc.f[i] = 0.f;
        const int n  = nb + (lane & 15);
        const int mb = (lane >> 4) * 8;
#pragma unroll 1
        for (int e = 0; e < NEXP; e++) {
            v8f c = {};
            c = bf16_gemm<HID>(&s.zbf[0][0], HID,
                               w_ex + e * HID * HID + nb * HID, HID, lane, c);
            const float be = b_ex[e * HID + n];
            VF u; u.v = c;
#pragma unroll
            for (int i = 0; i < 8; i++)
                acc.f[i] += s.gate[mb + i][e] * (u.f[i] + be);
        }
        if (mod == 0) acc_t = acc; else acc_i = acc;
        __syncthreads();   // gate/z/zbf/bufA reused by next phase
    }

    // ---- concat [t_out | i_out] as bf16 (reuse staging buffer) -------------
    __bf16* cat = s.bufA;   // [16][256]
    {
        const int n  = nb + (lane & 15);
        const int mb = (lane >> 4) * 8;
#pragma unroll
        for (int i = 0; i < 8; i++) {
            cat[(mb + i) * 256 + n]       = (__bf16)acc_t.f[i];
            cat[(mb + i) * 256 + HID + n] = (__bf16)acc_i.f[i];
        }
    }
    __syncthreads();

    // ---- fusion matmul [16x256] @ [256x128] --------------------------------
    {
        v8f c = {};
        c = bf16_gemm<2 * HID>(cat, 2 * HID, p.w_fus + nb * 2 * HID, 2 * HID, lane, c);
        store_tile_bias(&s.x[0][0], c, p.fus_b[nb + (lane & 15)], lane, nb);
    }
    __syncthreads();

    // fusion LayerNorm (eps 1e-5)
    if (tid < TILE) {
        float mean = 0.f;
        for (int h = 0; h < HID; h++) mean += s.x[tid][h];
        mean *= (1.f / HID);
        float var = 0.f;
        for (int h = 0; h < HID; h++) { float d = s.x[tid][h] - mean; var += d * d; }
        var *= (1.f / HID);
        const float rs = rsqrtf(var + 1e-5f);
        for (int h = 0; h < HID; h++)
            s.x[tid][h] = (s.x[tid][h] - mean) * rs * p.fus_ln_w[h] + p.fus_ln_b[h];
    }
    __syncthreads();

    // residual + relu + vectorized store (global_store_b128)
    for (int i = tid; i < TILE * HID / 4; i += 256) {
        const v4f sq = ((const v4f*)&s.seq[0][0])[i];
        v4f x = ((const v4f*)&s.x[0][0])[i];
        x.x = sq.x + fmaxf(x.x, 0.f);
        x.y = sq.y + fmaxf(x.y, 0.f);
        x.z = sq.z + fmaxf(x.z, 0.f);
        x.w = sq.w + fmaxf(x.w, 0.f);
        ((v4f*)p.out)[(long)t0 * (HID / 4) + i] = x;
    }
}

// ---------------------------------------------------------------------------
// Host launcher
// ---------------------------------------------------------------------------
extern "C" void kernel_launch(void* const* d_in, const int* in_sizes, int n_in,
                              void* d_out, int out_size, void* d_ws, size_t ws_size,
                              hipStream_t stream) {
    (void)in_sizes; (void)n_in; (void)out_size; (void)ws_size;

    // bf16 weight workspace layout (element offsets)
    __bf16* ws = (__bf16*)d_ws;
    size_t off = 0;
    __bf16* w_text = ws + off; off += (size_t)HID * DTX;        // fc_text_w
    __bf16* w_img  = ws + off; off += (size_t)HID * DIM;        // fc_img_w
    __bf16* w_mu_t = ws + off; off += (size_t)HID * HID;
    __bf16* w_sg_t = ws + off; off += (size_t)HID * HID;
    __bf16* w_mu_i = ws + off; off += (size_t)HID * HID;
    __bf16* w_sg_i = ws + off; off += (size_t)HID * HID;
    __bf16* w_texp = ws + off; off += (size_t)NEXP * HID * HID;
    __bf16* w_iexp = ws + off; off += (size_t)NEXP * HID * HID;
    __bf16* w_fus  = ws + off; off += (size_t)HID * 2 * HID;

    auto cvt = [&](int idx, __bf16* dst, int n) {
        const int n4 = n / 4;
        cvt_bf16_kernel<<<(n4 + 255) / 256, 256, 0, stream>>>(
            (const float*)d_in[idx], dst, n4);
    };
    cvt(7,  w_text, HID * DTX);
    cvt(9,  w_img,  HID * DIM);
    cvt(13, w_mu_t, HID * HID);
    cvt(15, w_sg_t, HID * HID);
    cvt(17, w_mu_i, HID * HID);
    cvt(19, w_sg_i, HID * HID);
    cvt(23, w_texp, NEXP * HID * HID);
    cvt(25, w_iexp, NEXP * HID * HID);
    cvt(27, w_fus,  HID * 2 * HID);

    Params p;
    p.ids        = (const int*)d_in[0];
    p.t_noise    = (const float*)d_in[1];
    p.i_noise    = (const float*)d_in[2];
    p.item_table = (const float*)d_in[3];
    p.pos_table  = (const float*)d_in[4];
    p.text_table = (const float*)d_in[5];
    p.img_table  = (const float*)d_in[6];
    p.fc_text_b  = (const float*)d_in[8];
    p.fc_img_b   = (const float*)d_in[10];
    p.ln_w       = (const float*)d_in[11];
    p.ln_b       = (const float*)d_in[12];
    p.mu_t_b     = (const float*)d_in[14];
    p.sg_t_b     = (const float*)d_in[16];
    p.mu_i_b     = (const float*)d_in[18];
    p.sg_i_b     = (const float*)d_in[20];
    p.gate_w     = (const float*)d_in[21];
    p.gate_b     = (const float*)d_in[22];
    p.texp_b     = (const float*)d_in[24];
    p.iexp_b     = (const float*)d_in[26];
    p.fus_b      = (const float*)d_in[28];
    p.fus_ln_w   = (const float*)d_in[29];
    p.fus_ln_b   = (const float*)d_in[30];
    p.w_text = w_text; p.w_img = w_img;
    p.w_mu_t = w_mu_t; p.w_sg_t = w_sg_t;
    p.w_mu_i = w_mu_i; p.w_sg_i = w_sg_i;
    p.w_texp = w_texp; p.w_iexp = w_iexp; p.w_fus = w_fus;
    p.out = (float*)d_out;

    fused_mm_moe_kernel<<<NTOK / TILE, 256, 0, stream>>>(p);
}
